// L_SA_2680059592808
// MI455X (gfx1250) — compile-verified, matrix operand order checked by reading, not run on hardware
//
#include <hip/hip_runtime.h>
#include <hip/hip_bf16.h>
#include <stdint.h>

// ---------------------------------------------------------------- types
typedef __bf16 bf16_t;
typedef __attribute__((ext_vector_type(16))) __bf16 v16bf;
typedef __attribute__((ext_vector_type(8)))  __bf16 v8bf;
typedef __attribute__((ext_vector_type(8)))  float  v8f;

#define DIMC   192
#define CH     96
#define NHB    4
#define HD     24
#define BATCH  16
#define IMH    128
#define IMW    128
#define LTOK   (IMH*IMW)
#define MTOT   (BATCH*LTOK)      /* 262144 */
#define NQKV   (3*DIMC)          /* 576 */

#if defined(__AMDGCN__) && __has_builtin(__builtin_amdgcn_tensor_load_to_lds)
#define USE_TDM 1
#endif
#if defined(__AMDGCN__) && __has_builtin(__builtin_amdgcn_cvt_pk_bf16_f32)
#define HAVE_PK_BF16 1
#endif

__device__ __forceinline__ bf16_t f2bf(float f) {
    union { float f; unsigned u; } x; x.f = f;
    unsigned r = x.u + 0x7FFFu + ((x.u >> 16) & 1u);   // round-to-nearest-even
    union { unsigned short s; __bf16 b; } y; y.s = (unsigned short)(r >> 16);
    return y.b;
}
__device__ __forceinline__ float bf2f(bf16_t b) {
    union { unsigned short s; __bf16 b; } y; y.b = b;
    union { unsigned u; float f; } x; x.u = ((unsigned)y.s) << 16;
    return x.f;
}

#ifdef USE_TDM
typedef __attribute__((ext_vector_type(4))) unsigned int u32x4;
typedef __attribute__((ext_vector_type(8))) int i32x8;
typedef __attribute__((ext_vector_type(4))) int i32x4;

// 2D TDM tile load: tile_x elems per row (data_size=2B), tile_y rows, global row
// stride row_stride_elems. LDS padding: after every 2^(pad_int_code+1) DWORDs,
// insert (pad_amt_code+1) DWORDs -> reproduces the 40-bf16 LDS pitch.
// D# field packing per CDNA5 ISA (groups 0/1; groups 2/3 zero => 2D).
// This toolchain exposes the 6-arg builtin:
//   (uint32x4 g0, int32x8 g1, int32x4 g2, int32x4 g3, int32x8 extra, i32 cpol)
__device__ __forceinline__ void tdm_load_tile_2d(unsigned lds_off, const void* gptr,
                                                 unsigned tile_x, unsigned tile_y,
                                                 unsigned row_stride_elems,
                                                 unsigned pad_int_code, unsigned pad_amt_code) {
    unsigned long long ga = (unsigned long long)(uintptr_t)gptr;
    u32x4 g0;
    g0[0] = 1u;                                   // count=1, user descriptor
    g0[1] = lds_off;                              // lds_addr (bytes)
    g0[2] = (unsigned)ga;                         // global_addr[31:0]
    g0[3] = (unsigned)(ga >> 32) | (2u << 30);    // global_addr[56:32] | type=2
    i32x8 g1;
    g1[0] = (int)((1u << 16)                      // data_size = 1 -> 2 bytes
                | (1u << 20)                      // pad_enable
                | (pad_int_code << 22)            // pad_interval
                | (pad_amt_code << 25));          // pad_amount
    g1[1] = (int)((row_stride_elems & 0xFFFFu) << 16);                 // tensor_dim0[15:0]
    g1[2] = (int)(((row_stride_elems >> 16) & 0xFFFFu)
                | ((tile_y & 0xFFFFu) << 16));                         // dim0 hi | dim1 lo
    g1[3] = (int)(((tile_y >> 16) & 0xFFFFu) | (tile_x << 16));        // dim1 hi | tile_dim0
    g1[4] = (int)(tile_y & 0xFFFFu);                                   // tile_dim1 | tile_dim2=0
    g1[5] = (int)row_stride_elems;                                     // dim0_stride[31:0]
    g1[6] = 0;
    g1[7] = 0;
    i32x4 z4 = {0, 0, 0, 0};
    i32x8 z8 = {0, 0, 0, 0, 0, 0, 0, 0};
    __builtin_amdgcn_tensor_load_to_lds(g0, g1, z4, z4, z8, 0);
}
#endif

// ------------------------------------------------- weight transpose (to N x K bf16)
__global__ void prep_weights(const float* __restrict__ qkv_w, const float* __restrict__ proj_w,
                             bf16_t* __restrict__ wT_qkv, bf16_t* __restrict__ wT_proj) {
    int idx = blockIdx.x * 256 + threadIdx.x;
    if (idx < NQKV * DIMC) {
        int n = idx / DIMC, k = idx % DIMC;
        wT_qkv[idx] = f2bf(qkv_w[k * NQKV + n]);
    }
    int idx2 = idx - NQKV * DIMC;
    if (idx2 >= 0 && idx2 < DIMC * DIMC) {
        int n = idx2 / DIMC, k = idx2 % DIMC;
        wT_proj[idx2] = f2bf(proj_w[k * DIMC + n]);
    }
}

// ------------------------------------------------- dynamic position bias -> rpb table
__global__ void dpb_kernel(const float* __restrict__ pos_proj_w, const float* __restrict__ pos_proj_b,
                           const float* __restrict__ ln_g, const float* __restrict__ ln_b,
                           const float* __restrict__ mlp_w, const float* __restrict__ mlp_b,
                           const float* __restrict__ head_w, const float* __restrict__ head_b,
                           float* __restrict__ rpb) {
    __shared__ float pos_s[21][NHB];
    int t = threadIdx.x;
    for (int bi = 0; bi < 2; ++bi) {
        int Hs = bi ? 4 : 2, Ws = bi ? 2 : 4;
        int R = (2 * Hs - 1) * (2 * Ws - 1);   // 21 for both branches
        if (t < R) {
            float x[24];
            float dh = (float)(1 - Hs + t / (2 * Ws - 1));
            float dw = (float)(1 - Ws + t % (2 * Ws - 1));
            const float* pw = pos_proj_w + bi * 2 * 24;
            const float* pb = pos_proj_b + bi * 24;
            for (int c = 0; c < 24; ++c) x[c] = dh * pw[c] + dw * pw[24 + c] + pb[c];
            for (int l = 0; l < 3; ++l) {
                float mu = 0.f;
                for (int c = 0; c < 24; ++c) mu += x[c];
                mu *= (1.f / 24.f);
                float var = 0.f;
                for (int c = 0; c < 24; ++c) { float d = x[c] - mu; var += d * d; }
                var *= (1.f / 24.f);
                float inv = rsqrtf(var + 1e-5f);
                const float* g = ln_g + (bi * 3 + l) * 24;
                const float* b = ln_b + (bi * 3 + l) * 24;
                float y[24];
                for (int c = 0; c < 24; ++c) {
                    float v = (x[c] - mu) * inv * g[c] + b[c];
                    y[c] = v > 0.f ? v : 0.f;
                }
                if (l < 2) {
                    const float* Wm = mlp_w + (bi * 2 + l) * 24 * 24;
                    const float* bm = mlp_b + (bi * 2 + l) * 24;
                    for (int o = 0; o < 24; ++o) {
                        float s = bm[o];
                        for (int c = 0; c < 24; ++c) s += y[c] * Wm[c * 24 + o];
                        x[o] = s;
                    }
                } else {
                    const float* Wh = head_w + bi * 24 * NHB;
                    const float* bh = head_b + bi * NHB;
                    for (int o = 0; o < NHB; ++o) {
                        float s = bh[o];
                        for (int c = 0; c < 24; ++c) s += y[c] * Wh[c * NHB + o];
                        pos_s[t][o] = s;
                    }
                }
            }
        }
        __syncthreads();
        for (int e = t; e < NHB * 8 * 8; e += blockDim.x) {
            int h = e >> 6, n = (e >> 3) & 7, m = e & 7;
            int in_ = n / Ws, jn = n % Ws, im = m / Ws, jm = m % Ws;
            int ridx = (in_ - im + Hs - 1) * (2 * Ws - 1) + (jn - jm) + Ws - 1;
            rpb[((bi * NHB + h) * 8 + n) * 8 + m] = pos_s[ridx][h];
        }
        __syncthreads();
    }
}

// ------------------------------------------------- WMMA GEMM: C[M,N] = A[M,K] * BT[N,K]^T (+bias)
// Block tile 128x64, BK=32, 256 threads = 8 waves, each wave a 32x32 sub-tile (2x2 WMMA).
// B tile staged by the Tensor Data Mover (one descriptor per K-step, wave 0 issues,
// TENSORcnt waited before the block barrier); A tile staged via VALU (fp32->bf16
// conversion for GEMM1) with v_cvt_pk_bf16_f32 when available.
template<bool A_F32, bool OUT_F32>
__global__ void wmma_gemm(const void* __restrict__ A_, const bf16_t* __restrict__ BT,
                          const float* __restrict__ bias, void* __restrict__ C_,
                          int K, int N) {
    __shared__ __align__(16) bf16_t As[128 * 40];   // pitch 40 bf16 = 80B (16B multiple)
    __shared__ __align__(16) bf16_t Bs[64 * 40];

    const int m0 = blockIdx.x * 128;
    const int n0 = blockIdx.y * 64;
    const int t = threadIdx.x;
    const int lane = t & 31, wv = t >> 5;
    const int wm = wv & 3, wn = wv >> 2;
    const int r = lane & 15, hi = lane >> 4;

    v8f acc[2][2];
    for (int i = 0; i < 2; ++i)
        for (int j = 0; j < 2; ++j)
            for (int e = 0; e < 8; ++e) acc[i][j][e] = 0.f;

    const int arow = t >> 1;            // 0..127
    const int acol = (t & 1) * 16;      // 0 or 16
    const int brow = t >> 2;            // 0..63
    const int bcol = (t & 3) * 8;       // 0,8,16,24

    for (int k0 = 0; k0 < K; k0 += 32) {
#ifdef USE_TDM
        // B tile 64(N) x 32(K) bf16 via TDM: pad 4 DWORDs after every 16 DWORDs
        // -> 40-bf16 LDS pitch. pad_interval code 3 (=16 DW), pad_amount code 3 (=4 DW).
        if (wv == 0) {
            tdm_load_tile_2d((unsigned)(uintptr_t)(void*)Bs,
                             BT + (size_t)n0 * K + k0,
                             /*tile_x=*/32, /*tile_y=*/64,
                             /*row_stride=*/(unsigned)K, 3, 3);
        }
#else
        {
            const v8bf* src = (const v8bf*)(BT + (long)(n0 + brow) * K + k0 + bcol);
            *(v8bf*)(Bs + brow * 40 + bcol) = *src;
        }
#endif
        // stage A tile
        if (A_F32) {
            const float* A = (const float*)A_;
            const float* src = A + (long)(m0 + arow) * K + k0 + acol;
            bf16_t* dst = As + arow * 40 + acol;
#ifdef HAVE_PK_BF16
            #pragma unroll
            for (int e = 0; e < 16; e += 2) {
                auto p = __builtin_amdgcn_cvt_pk_bf16_f32(src[e], src[e + 1]);
                __builtin_memcpy(dst + e, &p, 4);
            }
#else
            #pragma unroll
            for (int e = 0; e < 16; ++e) dst[e] = f2bf(src[e]);
#endif
            if (k0 + 32 < K) __builtin_prefetch(src + 32, 0, 3);   // global_prefetch_b8
        } else {
            const bf16_t* A = (const bf16_t*)A_;
            const v8bf* src = (const v8bf*)(A + (long)(m0 + arow) * K + k0 + acol);
            v8bf* dst = (v8bf*)(As + arow * 40 + acol);
            dst[0] = src[0];
            dst[1] = src[1];
            if (k0 + 32 < K) __builtin_prefetch((const bf16_t*)src + 32, 0, 3);
        }
#ifdef USE_TDM
        if (wv == 0) __builtin_amdgcn_s_wait_tensorcnt(0);
#endif
        __syncthreads();

        // B fragments for this wave's two N sub-tiles
        v16bf bfrag[2];
        #pragma unroll
        for (int ni = 0; ni < 2; ++ni) {
            const bf16_t* pb = Bs + (wn * 32 + ni * 16 + r) * 40 + hi * 8;
            v8bf blo = *(const v8bf*)pb;         // K 0..7  (or 8..15  for hi lanes)
            v8bf bhi = *(const v8bf*)(pb + 16);  // K 16..23 (or 24..31 for hi lanes)
            bfrag[ni] = __builtin_shufflevector(blo, bhi, 0,1,2,3,4,5,6,7,8,9,10,11,12,13,14,15);
        }
        #pragma unroll
        for (int mi = 0; mi < 2; ++mi) {
            const bf16_t* pa = As + (wm * 32 + mi * 16 + r) * 40 + hi * 8;
            v8bf alo = *(const v8bf*)pa;
            v8bf ahi = *(const v8bf*)(pa + 16);
            v16bf a = __builtin_shufflevector(alo, ahi, 0,1,2,3,4,5,6,7,8,9,10,11,12,13,14,15);
            #pragma unroll
            for (int ni = 0; ni < 2; ++ni) {
                acc[mi][ni] = __builtin_amdgcn_wmma_f32_16x16x32_bf16(
                    false, a, false, bfrag[ni], (short)0, acc[mi][ni], false, false);
            }
        }
        __syncthreads();
    }

    // store: VGPR j of C/D holds rows M=j (lanes 0-15) and M=j+8 (lanes 16-31)
    const long crow_base = m0 + wm * 32;
    const int  ccol_base = n0 + wn * 32;
    #pragma unroll
    for (int mi = 0; mi < 2; ++mi)
        #pragma unroll
        for (int ni = 0; ni < 2; ++ni) {
            int col = ccol_base + ni * 16 + r;
            long rowb = crow_base + mi * 16 + hi * 8;
            if (OUT_F32) {
                float* C = (float*)C_;
                float bv = bias ? bias[col] : 0.f;
                #pragma unroll
                for (int j = 0; j < 8; ++j) C[(rowb + j) * N + col] = acc[mi][ni][j] + bv;
            } else {
                bf16_t* C = (bf16_t*)C_;
                #pragma unroll
                for (int j = 0; j < 8; ++j) C[(rowb + j) * N + col] = f2bf(acc[mi][ni][j]);
            }
        }
}

// ------------------------------------------------- fused shifted-window attention
// One window (N=8, 4 heads) per wave32: lane = head*8 + row. Roll is folded into the
// index remap: rolled (h',w') maps to pixel ((h'+sh0)%128,(w'+sh1)%128) for both the
// qkv gather and the output scatter.
__global__ void attn_kernel(const bf16_t* __restrict__ qkv, const float* __restrict__ rpb,
                            bf16_t* __restrict__ act) {
    __shared__ bf16_t kk[8][8][96];
    __shared__ bf16_t vv[8][8][96];
    const int bi = blockIdx.y;
    const int Hs = bi ? 4 : 2, Ws = bi ? 2 : 4;
    const int sh0 = bi ? 2 : 1, sh1 = bi ? 1 : 2;
    const int wave = threadIdx.x >> 5, lane = threadIdx.x & 31;
    const long wid = (long)blockIdx.x * 8 + wave;
    const int nWx = IMW / Ws;
    const int nW = (IMH / Hs) * nWx;   // 2048
    const int b  = (int)(wid / nW);
    const int wr = (int)(wid % nW);
    const int wh = wr / nWx, wwi = wr % nWx;

    // cooperative stage of K and V for this wave's window (8 elems x 96 ch)
    #pragma unroll
    for (int it = 0; it < 3; ++it) {
        int chunk = it * 32 + lane;            // 0..95
        int elem = chunk / 12, cp = chunk % 12;
        int ie = elem / Ws, je = elem % Ws;
        int hp = wh * Hs + ie, wp = wwi * Ws + je;
        int ph = (hp + sh0) & 127, pw = (wp + sh1) & 127;
        long m = ((long)b << 14) + ph * 128 + pw;
        const v8bf* kp = (const v8bf*)(qkv + m * NQKV + DIMC     + bi * CH + cp * 8);
        const v8bf* vp = (const v8bf*)(qkv + m * NQKV + 2 * DIMC + bi * CH + cp * 8);
        *(v8bf*)(&kk[wave][elem][cp * 8]) = *kp;
        *(v8bf*)(&vv[wave][elem][cp * 8]) = *vp;
    }
    __syncthreads();

    const int h = lane >> 3, n = lane & 7;
    int in_ = n / Ws, jn = n % Ws;
    int hp = wh * Hs + in_, wp = wwi * Ws + jn;
    int ph = (hp + sh0) & 127, pw = (wp + sh1) & 127;
    long mq = ((long)b << 14) + ph * 128 + pw;
    int rhn = hp < (IMH - Hs) ? 0 : (hp < (IMH - sh0) ? 1 : 2);
    int rwn = wp < (IMW - Ws) ? 0 : (wp < (IMW - sh1) ? 1 : 2);
    int ridn = rhn * 3 + rwn;

    float q[HD];
    {
        const bf16_t* qp = qkv + mq * NQKV + bi * CH + h * HD;
        #pragma unroll
        for (int c = 0; c < HD; ++c) q[c] = bf2f(qp[c]);
    }

    const float scale = 0.20412414523193154f;   // 24^-0.5
    float logit[8];
    float mx = -1e30f;
    #pragma unroll
    for (int jj = 0; jj < 8; ++jj) {
        float d = 0.f;
        const bf16_t* kr = &kk[wave][jj][h * HD];
        #pragma unroll
        for (int c = 0; c < HD; ++c) d += q[c] * bf2f(kr[c]);
        int ij = jj / Ws, jc = jj % Ws;
        int hpj = wh * Hs + ij, wpj = wwi * Ws + jc;
        int rhj = hpj < (IMH - Hs) ? 0 : (hpj < (IMH - sh0) ? 1 : 2);
        int rwj = wpj < (IMW - Ws) ? 0 : (wpj < (IMW - sh1) ? 1 : 2);
        float mask = (rhj * 3 + rwj) == ridn ? 0.f : -100.f;
        float lv = d * scale + rpb[((bi * NHB + h) * 8 + n) * 8 + jj] + mask;
        logit[jj] = lv;
        mx = fmaxf(mx, lv);
    }
    float sum = 0.f;
    #pragma unroll
    for (int jj = 0; jj < 8; ++jj) { logit[jj] = __expf(logit[jj] - mx); sum += logit[jj]; }
    float inv = 1.f / sum;
    float out[HD];
    #pragma unroll
    for (int c = 0; c < HD; ++c) out[c] = 0.f;
    #pragma unroll
    for (int jj = 0; jj < 8; ++jj) {
        float p = logit[jj] * inv;
        const bf16_t* vr = &vv[wave][jj][h * HD];
        #pragma unroll
        for (int c = 0; c < HD; ++c) out[c] += p * bf2f(vr[c]);
    }
    bf16_t* op = act + mq * DIMC + bi * CH + h * HD;
    #pragma unroll
    for (int c = 0; c < HD; ++c) op[c] = f2bf(out[c]);
}

// ------------------------------------------------- depthwise 3x3 conv (SAME) + add into act
__global__ void conv_add_kernel(const bf16_t* __restrict__ qkv, const float* __restrict__ conv_w,
                                const float* __restrict__ conv_b, bf16_t* __restrict__ act) {
    long idx = (long)blockIdx.x * 256 + threadIdx.x;
    if (idx >= (long)MTOT * DIMC) return;
    int c = (int)(idx % DIMC);
    long pix = idx / DIMC;
    int b = (int)(pix >> 14);
    int hw = (int)(pix & 16383);
    int h_ = hw >> 7, w_ = hw & 127;
    float s = conv_b[c];
    #pragma unroll
    for (int dy = 0; dy < 3; ++dy) {
        int hh = h_ + dy - 1;
        if (hh < 0 || hh >= IMH) continue;
        #pragma unroll
        for (int dx = 0; dx < 3; ++dx) {
            int wc = w_ + dx - 1;
            if (wc < 0 || wc >= IMW) continue;
            long mm = ((long)b << 14) + hh * 128 + wc;
            s += bf2f(qkv[mm * NQKV + 2 * DIMC + c]) * conv_w[c * 9 + dy * 3 + dx];
        }
    }
    act[idx] = f2bf(bf2f(act[idx]) + s);
}

// ------------------------------------------------- launch
extern "C" void kernel_launch(void* const* d_in, const int* in_sizes, int n_in,
                              void* d_out, int out_size, void* d_ws, size_t ws_size,
                              hipStream_t stream) {
    const float* x          = (const float*)d_in[0];
    const float* qkv_w      = (const float*)d_in[1];
    const float* proj_w     = (const float*)d_in[2];
    const float* proj_b     = (const float*)d_in[3];
    const float* conv_w     = (const float*)d_in[4];
    const float* conv_b     = (const float*)d_in[5];
    const float* pos_proj_w = (const float*)d_in[6];
    const float* pos_proj_b = (const float*)d_in[7];
    const float* ln_g       = (const float*)d_in[8];
    const float* ln_b       = (const float*)d_in[9];
    const float* mlp_w      = (const float*)d_in[10];
    const float* mlp_b      = (const float*)d_in[11];
    const float* head_w     = (const float*)d_in[12];
    const float* head_b     = (const float*)d_in[13];

    char* ws = (char*)d_ws;
    size_t off = 0;
    auto alloc = [&](size_t bytes) -> void* {
        void* p = ws + off;
        off = (off + bytes + 255) & ~(size_t)255;
        return p;
    };
    bf16_t* wT_qkv  = (bf16_t*)alloc((size_t)NQKV * DIMC * 2);
    bf16_t* wT_proj = (bf16_t*)alloc((size_t)DIMC * DIMC * 2);
    float*  rpb     = (float*) alloc((size_t)2 * NHB * 8 * 8 * 4);
    bf16_t* qkv     = (bf16_t*)alloc((size_t)MTOT * NQKV * 2);   // ~302 MB
    bf16_t* act     = (bf16_t*)alloc((size_t)MTOT * DIMC * 2);   // ~101 MB
    (void)ws_size;

    prep_weights<<<((NQKV * DIMC + DIMC * DIMC) + 255) / 256, 256, 0, stream>>>(
        qkv_w, proj_w, wT_qkv, wT_proj);
    dpb_kernel<<<1, 64, 0, stream>>>(pos_proj_w, pos_proj_b, ln_g, ln_b,
                                     mlp_w, mlp_b, head_w, head_b, rpb);
    // qkv = x @ qkv_w   (fp32 A converted to bf16 in LDS staging)
    wmma_gemm<true, false><<<dim3(MTOT / 128, NQKV / 64), 256, 0, stream>>>(
        (const void*)x, wT_qkv, nullptr, (void*)qkv, DIMC, NQKV);
    // windowed attention (both branches)
    attn_kernel<<<dim3((BATCH * 2048) / 8, 2), 256, 0, stream>>>(qkv, rpb, act);
    // act += depthwise conv(v)
    conv_add_kernel<<<(unsigned)(((long)MTOT * DIMC + 255) / 256), 256, 0, stream>>>(
        qkv, conv_w, conv_b, act);
    // out = act @ proj_w + proj_b  (fp32 out)
    wmma_gemm<false, true><<<dim3(MTOT / 128, DIMC / 64), 256, 0, stream>>>(
        (const void*)act, wT_proj, proj_b, d_out, DIMC, DIMC);
}
